// Qwen3_5GatedDeltaNet_56186762166686
// MI455X (gfx1250) — matrix operand order repaired and checked by reading, not compiled, not run on hardware
//
#include <hip/hip_runtime.h>
#include <hip/hip_bf16.h>
#include <math.h>

// ---------------- problem constants ----------------
#define BB   2
#define TT   2048
#define DD   1024
#define HVn  16
#define HKn  8
#define DKn  64
#define DVn  64
#define KDIM 512
#define VDIM 1024
#define CONV 2048
#define KSz  4
#define EPSf 1e-6f
#define MM   (BB*TT)          // 4096 rows

typedef __attribute__((ext_vector_type(16))) __bf16 v16bf;
typedef __attribute__((ext_vector_type(8)))  float  v8f;
typedef __attribute__((__vector_size__(16))) int vi4;          // 16B chunk
typedef __attribute__((address_space(3))) vi4 lds_vi4;         // LDS 16B chunk

#if defined(__has_builtin)
#  if __has_builtin(__builtin_amdgcn_global_load_async_to_lds_b128)
#    define HAS_ASYNC_LDS 1
#  else
#    define HAS_ASYNC_LDS 0
#  endif
#else
#  define HAS_ASYNC_LDS 0
#endif

__device__ __forceinline__ unsigned short f2bf(float f) {
    union { float f; unsigned u; } x; x.f = f;
    unsigned r = x.u + 0x7FFFu + ((x.u >> 16) & 1u);   // RNE
    return (unsigned short)(r >> 16);
}

__device__ __forceinline__ void wait_async_lds() {
#if HAS_ASYNC_LDS
#  if __has_builtin(__builtin_amdgcn_s_wait_asynccnt)
    __builtin_amdgcn_s_wait_asynccnt(0);
#  else
    asm volatile("s_wait_asynccnt 0x0" ::: "memory");
#  endif
#endif
}

// ---------------- fp32 -> bf16 conversion ----------------
__global__ void cvt_bf16_kernel(const float* __restrict__ src,
                                unsigned short* __restrict__ dst, int n) {
    int i = blockIdx.x * blockDim.x + threadIdx.x;
    if (i < n) dst[i] = f2bf(src[i]);
}

// ---------------- WMMA GEMM: C[M,N] = A[M,K] @ B[K,N] ----------------
// A,B bf16 row-major; C fp32 row-major.
// Block = 256 threads (8 waves) -> 128x128 C tile; wave -> 64x32 sub-tile
// (4x2 WMMA accumulators). A staged row-major in LDS (pitch 40 bf16),
// B staged TRANSPOSED in LDS (pitch 40) so B fragments are contiguous.
#define APITCH 40
__global__ void __launch_bounds__(256)
wmma_gemm_bf16(const unsigned short* __restrict__ A,
               const unsigned short* __restrict__ Bm,
               float* __restrict__ C,
               int Mi, int Ni, int Ki) {
    __shared__ __align__(16) unsigned short As[128 * APITCH]; // 128 rows x 32 K
    __shared__ __align__(16) unsigned short Bs[128 * APITCH]; // 128 cols x 32 K (transposed)

    const int tid  = threadIdx.x;
    const int lane = tid & 31;
    const int l16  = lane & 15;
    const int h    = lane >> 4;
    const int wave = tid >> 5;        // 0..7
    const int wm   = wave & 1;        // M strip (64 rows)
    const int wn   = wave >> 1;       // N strip (32 cols)

    const int rowBase = blockIdx.y * 128;
    const int colBase = blockIdx.x * 128;

    v8f acc[4][2];
#pragma unroll
    for (int i = 0; i < 4; ++i)
#pragma unroll
        for (int j = 0; j < 2; ++j) acc[i][j] = (v8f){};

    for (int kb = 0; kb < Ki; kb += 32) {
        __syncthreads();   // previous compute finished reading LDS

        // ---- stage A: 128x32 bf16, 16B chunks, layout-compatible copy ----
#pragma unroll
        for (int c = tid; c < 512; c += 256) {
            int row = c >> 2, kc = (c & 3) * 8;
            const unsigned short* gsrc =
                A + (size_t)(rowBase + row) * Ki + kb + kc;
            unsigned short* ldst = &As[row * APITCH + kc];
#if HAS_ASYNC_LDS
            __builtin_amdgcn_global_load_async_to_lds_b128(
                (vi4*)gsrc, (lds_vi4*)ldst, 0, 0);
#else
            *(uint4*)ldst = *(const uint4*)gsrc;
#endif
        }

        // ---- stage B transposed: read 8 cols coalesced, scatter to LDS ----
#pragma unroll
        for (int c = tid; c < 512; c += 256) {
            int kr = c >> 4, nc = (c & 15) * 8;
            union { uint4 q; unsigned short u[8]; } vld;
            vld.q = *(const uint4*)(Bm + (size_t)(kb + kr) * Ni + colBase + nc);
#pragma unroll
            for (int e = 0; e < 8; ++e)
                Bs[(nc + e) * APITCH + kr] = vld.u[e];
        }

        wait_async_lds();
        __syncthreads();   // LDS tiles ready

        // ---- fragments + 8 WMMAs ----
        union F { v16bf v; uint4 q[2]; };
        F af[4], bf[2];
#pragma unroll
        for (int mt = 0; mt < 4; ++mt) {
            int row = wm * 64 + mt * 16 + l16;
            af[mt].q[0] = *(const uint4*)&As[row * APITCH + 8 * h];      // K=8h..8h+7
            af[mt].q[1] = *(const uint4*)&As[row * APITCH + 16 + 8 * h]; // K=16+8h..
        }
#pragma unroll
        for (int nt = 0; nt < 2; ++nt) {
            int col = wn * 32 + nt * 16 + l16;
            bf[nt].q[0] = *(const uint4*)&Bs[col * APITCH + 8 * h];      // K=8h..8h+7
            bf[nt].q[1] = *(const uint4*)&Bs[col * APITCH + 16 + 8 * h]; // K=16+8h..
        }
#pragma unroll
        for (int mt = 0; mt < 4; ++mt)
#pragma unroll
            for (int nt = 0; nt < 2; ++nt)
                acc[mt][nt] = __builtin_amdgcn_wmma_f32_16x16x32_bf16(
                    false, af[mt].v, false, bf[nt].v,
                    (short)0, acc[mt][nt], false, false);
    }

#pragma unroll
    for (int mt = 0; mt < 4; ++mt)
#pragma unroll
        for (int nt = 0; nt < 2; ++nt)
#pragma unroll
            for (int r = 0; r < 8; ++r) {
                int cr = rowBase + wm * 64 + mt * 16 + 8 * h + r;
                int cc = colBase + wn * 32 + nt * 16 + l16;
                C[(size_t)cr * Ni + cc] = acc[mt][nt][r];
            }
}

// ---------------- beta / g small projections ----------------
__global__ void small_proj_kernel(const float* __restrict__ hs,
                                  const float* __restrict__ Wb,
                                  const float* __restrict__ Wa,
                                  const float* __restrict__ dtb,
                                  const float* __restrict__ Alog,
                                  float* __restrict__ beta,
                                  float* __restrict__ g) {
    const int m   = blockIdx.x;
    const int tid = threadIdx.x;          // 256
    const int hh  = tid & 15;
    const int s   = tid >> 4;
    __shared__ float hrow[DD];
    __shared__ float pb[HVn][16];
    __shared__ float pa[HVn][16];

    for (int i = tid; i < DD; i += 256) hrow[i] = hs[(size_t)m * DD + i];
    __syncthreads();

    float sb = 0.f, sa = 0.f;
    for (int k = s; k < DD; k += 16) {
        float x = hrow[k];
        sb += x * Wb[(size_t)k * HVn + hh];
        sa += x * Wa[(size_t)k * HVn + hh];
    }
    pb[hh][s] = sb; pa[hh][s] = sa;
    __syncthreads();

    if (s == 0) {
        float tb = 0.f, ta = 0.f;
#pragma unroll
        for (int j = 0; j < 16; ++j) { tb += pb[hh][j]; ta += pa[hh][j]; }
        beta[(size_t)m * HVn + hh] = 1.f / (1.f + __expf(-tb));
        float x  = ta + dtb[hh];
        float sp = (x > 20.f) ? x : log1pf(__expf(x));
        g[(size_t)m * HVn + hh] = -__expf(Alog[hh]) * sp;
    }
}

// ---------------- causal conv(KS=4) + SiLU + split + l2norm ----------------
__global__ void conv_silu_split_kernel(const float* __restrict__ mixed,
                                       const float* __restrict__ cw,
                                       float* __restrict__ qb,
                                       float* __restrict__ kbuf,
                                       float* __restrict__ vb) {
    const int m   = blockIdx.x;
    const int b   = m / TT;
    const int t   = m % TT;
    const int tid = threadIdx.x;          // 256
    __shared__ float buf[CONV];
    __shared__ float part[16][16];
    __shared__ float rs[16];

    for (int c = tid; c < CONV; c += 256) {
        float acc = 0.f;
#pragma unroll
        for (int i = 0; i < KSz; ++i) {
            int tt = t - (KSz - 1) + i;
            if (tt >= 0)
                acc += mixed[((size_t)b * TT + tt) * CONV + c] * cw[(size_t)c * KSz + i];
        }
        buf[c] = acc / (1.f + __expf(-acc));      // SiLU
    }
    __syncthreads();

    {
        int grp = tid >> 4, l = tid & 15;
        float s = 0.f;
#pragma unroll
        for (int j = 0; j < 4; ++j) {
            float x = buf[grp * 64 + l * 4 + j];
            s += x * x;
        }
        part[grp][l] = s;
    }
    __syncthreads();
    if ((tid & 15) == 0) {
        int grp = tid >> 4;
        float s = 0.f;
#pragma unroll
        for (int j = 0; j < 16; ++j) s += part[grp][j];
        rs[grp] = rsqrtf(s + EPSf);
    }
    __syncthreads();

    for (int c = tid; c < CONV; c += 256) {
        float val = buf[c];
        if (c < KDIM) {
            qb[(size_t)m * KDIM + c] = val * rs[c >> 6] * 0.125f;   // * DK^-0.5
        } else if (c < 2 * KDIM) {
            kbuf[(size_t)m * KDIM + (c - KDIM)] = val * rs[c >> 6];
        } else {
            vb[(size_t)m * VDIM + (c - 2 * KDIM)] = val;
        }
    }
}

// ---------------- gated delta-rule recurrence ----------------
__global__ void recurrence_kernel(const float* __restrict__ qb,
                                  const float* __restrict__ kb,
                                  const float* __restrict__ vb,
                                  const float* __restrict__ gb,
                                  const float* __restrict__ bbuf,
                                  float* __restrict__ ob) {
    const int bh  = blockIdx.x;           // 0..31
    const int b   = bh >> 4;
    const int hv  = bh & 15;
    const int hk  = hv >> 1;              // GQA rep = 2
    const int tid = threadIdx.x;          // 256
    const int v   = tid & 63;
    const int kg  = tid >> 6;
    const int k0  = kg * 16;

    __shared__ float kS[DKn], qS[DKn], vS[DVn];
    __shared__ float red[4][DVn];

    float S[16];
#pragma unroll
    for (int i = 0; i < 16; ++i) S[i] = 0.f;

    for (int t = 0; t < TT; ++t) {
        const size_t m = (size_t)b * TT + t;
        if (tid < 64) {
            kS[tid] = kb[m * KDIM + hk * DKn + tid];
            qS[tid] = qb[m * KDIM + hk * DKn + tid];
            vS[tid] = vb[m * VDIM + hv * DVn + tid];
        }
        const float ge = __expf(gb[m * HVn + hv]);
        const float bt = bbuf[m * HVn + hv];
        __syncthreads();

        float part = 0.f;
#pragma unroll
        for (int i = 0; i < 16; ++i) { S[i] *= ge; part += kS[k0 + i] * S[i]; }
        red[kg][v] = part;
        __syncthreads();
        const float kv    = red[0][v] + red[1][v] + red[2][v] + red[3][v];
        const float delta = (vS[v] - kv) * bt;
        __syncthreads();

        float op = 0.f;
#pragma unroll
        for (int i = 0; i < 16; ++i) {
            float kk = kS[k0 + i];
            S[i] += kk * delta;
            op   += qS[k0 + i] * S[i];
        }
        red[kg][v] = op;
        __syncthreads();
        if (tid < 64)
            ob[m * VDIM + hv * DVn + v] =
                red[0][v] + red[1][v] + red[2][v] + red[3][v];
        __syncthreads();
    }
}

// ---------------- gated RMSNorm -> bf16 for out-proj ----------------
__global__ void gated_norm_kernel(const float* __restrict__ ob,
                                  const float* __restrict__ zb,
                                  const float* __restrict__ nw,
                                  unsigned short* __restrict__ y) {
    const int m   = blockIdx.x;
    const int tid = threadIdx.x;          // 256
    const int hv  = tid >> 4;
    const int l   = tid & 15;
    __shared__ float part[HVn][16];
    __shared__ float rs[HVn];

    float s = 0.f;
#pragma unroll
    for (int j = 0; j < 4; ++j) {
        float o = ob[(size_t)m * VDIM + hv * DVn + l * 4 + j];
        s += o * o;
    }
    part[hv][l] = s;
    __syncthreads();
    if (l == 0) {
        float tot = 0.f;
#pragma unroll
        for (int j = 0; j < 16; ++j) tot += part[hv][j];
        rs[hv] = rsqrtf(tot * (1.f / 64.f) + EPSf);
    }
    __syncthreads();
    const float r = rs[hv];
#pragma unroll
    for (int j = 0; j < 4; ++j) {
        int dv = l * 4 + j;
        size_t idx = (size_t)m * VDIM + hv * DVn + dv;
        float o = ob[idx] * r * nw[dv];
        float z = zb[idx];
        float sil = z / (1.f + __expf(-z));
        y[idx] = f2bf(o * sil);
    }
}

// ---------------- launcher ----------------
extern "C" void kernel_launch(void* const* d_in, const int* in_sizes, int n_in,
                              void* d_out, int out_size, void* d_ws, size_t ws_size,
                              hipStream_t stream) {
    const float* hs    = (const float*)d_in[0];
    const float* Wqkv  = (const float*)d_in[1];
    const float* Wz    = (const float*)d_in[2];
    const float* Wb    = (const float*)d_in[3];
    const float* Wa    = (const float*)d_in[4];
    const float* dtb   = (const float*)d_in[5];
    const float* Alog  = (const float*)d_in[6];
    const float* convw = (const float*)d_in[7];
    const float* normw = (const float*)d_in[8];
    const float* Wout  = (const float*)d_in[9];
    float* out = (float*)d_out;

    char* w = (char*)d_ws;
    size_t off = 0;
    auto alloc = [&](size_t bytes) -> void* {
        void* p = w + off;
        off += (bytes + 255) & ~(size_t)255;
        return p;
    };
    unsigned short* hbf    = (unsigned short*)alloc((size_t)MM * DD * 2);
    unsigned short* wqkvbf = (unsigned short*)alloc((size_t)DD * CONV * 2);
    unsigned short* wzbf   = (unsigned short*)alloc((size_t)DD * VDIM * 2);
    unsigned short* woutbf = (unsigned short*)alloc((size_t)VDIM * DD * 2);
    float* mixed = (float*)alloc((size_t)MM * CONV * 4);
    float* zbuf  = (float*)alloc((size_t)MM * VDIM * 4);
    float* betab = (float*)alloc((size_t)MM * HVn * 4);
    float* gbuf  = (float*)alloc((size_t)MM * HVn * 4);
    float* qb    = (float*)alloc((size_t)MM * KDIM * 4);
    float* kbuf  = (float*)alloc((size_t)MM * KDIM * 4);
    float* vb    = (float*)alloc((size_t)MM * VDIM * 4);
    float* ob    = (float*)alloc((size_t)MM * VDIM * 4);
    unsigned short* ybf = (unsigned short*)alloc((size_t)MM * VDIM * 2);

    {
        int n;
        n = MM * DD;
        cvt_bf16_kernel<<<(n + 255) / 256, 256, 0, stream>>>(hs, hbf, n);
        n = DD * CONV;
        cvt_bf16_kernel<<<(n + 255) / 256, 256, 0, stream>>>(Wqkv, wqkvbf, n);
        n = DD * VDIM;
        cvt_bf16_kernel<<<(n + 255) / 256, 256, 0, stream>>>(Wz, wzbf, n);
        n = VDIM * DD;
        cvt_bf16_kernel<<<(n + 255) / 256, 256, 0, stream>>>(Wout, woutbf, n);
    }

    // mixed_qkv = H @ W_qkv   [4096 x 2048]
    wmma_gemm_bf16<<<dim3(CONV / 128, MM / 128), 256, 0, stream>>>(
        hbf, wqkvbf, mixed, MM, CONV, DD);

    // z = H @ W_z             [4096 x 1024]
    wmma_gemm_bf16<<<dim3(VDIM / 128, MM / 128), 256, 0, stream>>>(
        hbf, wzbf, zbuf, MM, VDIM, DD);

    small_proj_kernel<<<MM, 256, 0, stream>>>(hs, Wb, Wa, dtb, Alog, betab, gbuf);

    conv_silu_split_kernel<<<MM, 256, 0, stream>>>(mixed, convw, qb, kbuf, vb);

    recurrence_kernel<<<BB * HVn, 256, 0, stream>>>(qb, kbuf, vb, gbuf, betab, ob);

    gated_norm_kernel<<<MM, 256, 0, stream>>>(ob, zbuf, normw, ybf);

    // out = y @ W_out         [4096 x 1024]
    wmma_gemm_bf16<<<dim3(DD / 128, MM / 128), 256, 0, stream>>>(
        ybf, woutbf, out, MM, DD, VDIM);
}
